// ProEncoder_88364657148486
// MI455X (gfx1250) — compile-verified, matrix operand order checked by reading, not run on hardware
//
#include <hip/hip_runtime.h>
#include <hip/hip_bf16.h>
#include <math.h>

// ---------------------------------------------------------------------------
// Model constants (from reference)
// ---------------------------------------------------------------------------
#define B_    16
#define S_    512
#define D_    512
#define FF_   1024
#define L_    5
#define SPAN_ 10

typedef __attribute__((ext_vector_type(16))) __bf16 v16bf;
typedef __attribute__((ext_vector_type(8)))  float  v8f;

// ---------------------------------------------------------------------------
// bf16 helpers (raw ushort bits)
// ---------------------------------------------------------------------------
__device__ __forceinline__ unsigned short f2bf(float f) {
    unsigned u = __float_as_uint(f);
    unsigned r = u + 0x7FFFu + ((u >> 16) & 1u);   // round-to-nearest-even
    return (unsigned short)(r >> 16);
}

union Frag16 { uint4 u[2]; v16bf v; };

// WMMA 16-bit operand fragment (A layout; B identical on the N index since we
// keep B transposed as Bt[N][K]).
// lanes 0-15: row = lane, K chunks {0..7, 16..23}
// lanes 16-31: row = lane-16, K chunks {8..15, 24..31}
__device__ __forceinline__ v16bf load_frag(const unsigned short* p) {
    Frag16 f;
    f.u[0] = *(const uint4*)(p);
    f.u[1] = *(const uint4*)(p + 16);
    return f.v;
}

// ---------------------------------------------------------------------------
// CDNA5 async global->LDS DMA (ASYNCcnt-tracked, no VGPR bounce)
// ---------------------------------------------------------------------------
__device__ __forceinline__ unsigned lds_byte_addr(const void* p) {
    // generic -> addrspace(3): hardware adds the wave's LDS base itself
    return (unsigned)(uintptr_t)(const __attribute__((address_space(3))) char*)p;
}

__device__ __forceinline__ void async_copy16(unsigned lds_off,
                                             const unsigned short* gsrc) {
    asm volatile("global_load_async_to_lds_b128 %0, %1, off"
                 :: "v"(lds_off), "v"(gsrc) : "memory");
}

__device__ __forceinline__ void wait_async() {
    asm volatile("s_wait_asynccnt 0x0" ::: "memory");
}

// ---------------------------------------------------------------------------
// Shared WMMA core: 128x128 block tile, 256 threads (8 waves).
// Wave (wm,wn) grid 4x2: each wave computes 32x64 (2 A-frags x 4 B-frags,
// 8 WMMAs per K-step).  Async double-buffered LDS staging:
// stage = A(128x32)=8KB + B(128x32)=8KB; smem = 2*8192 ushorts (32KB).
// ---------------------------------------------------------------------------
__device__ __forceinline__ void wmma_tile_128x128(
    const unsigned short* __restrict__ A,  int lda, int M, int m0,
    const unsigned short* __restrict__ Bt, int ldb, int N, int n0,
    int K, unsigned short* smem, v8f acc[8])
{
    const int tid  = threadIdx.x;
    const int wave = tid >> 5;
    const int lane = tid & 31;
    const int lr   = lane & 15;
    const int kb   = (lane < 16) ? 0 : 8;
    const int wm   = (wave & 3) * 32;   // 4 row strips of 32
    const int wn   = (wave >> 2) * 64;  // 2 col halves of 64

    const unsigned smem_b = lds_byte_addr(smem);

    // A: 128 rows x 64B = 512 chunks of 16B -> 2 per thread (rows r, r+64)
    // B: 128 rows x 64B = 512 chunks of 16B -> 2 per thread (rows r, r+64)
    const int ar = tid >> 2, aq = tid & 3;           // row 0..63, 16B quarter
    int gmA0 = m0 + ar;      if (gmA0 > M - 1) gmA0 = M - 1;
    int gmA1 = m0 + ar + 64; if (gmA1 > M - 1) gmA1 = M - 1;
    int gnB0 = n0 + ar;      if (gnB0 > N - 1) gnB0 = N - 1;
    int gnB1 = n0 + ar + 64; if (gnB1 > N - 1) gnB1 = N - 1;

    const unsigned short* a0 = A  + (size_t)gmA0 * lda + aq * 8;
    const unsigned short* a1 = A  + (size_t)gmA1 * lda + aq * 8;
    const unsigned short* b0 = Bt + (size_t)gnB0 * ldb + aq * 8;
    const unsigned short* b1 = Bt + (size_t)gnB1 * ldb + aq * 8;
    const unsigned off0 = (unsigned)(ar * 64 + aq * 16);
    const unsigned off1 = (unsigned)((ar + 64) * 64 + aq * 16);

    auto issue = [&](int k0, int buf) {
        unsigned as_b = smem_b + (unsigned)buf * 16384u;
        unsigned bs_b = as_b + 8192u;
        async_copy16(as_b + off0, a0 + k0);
        async_copy16(as_b + off1, a1 + k0);
        async_copy16(bs_b + off0, b0 + k0);
        async_copy16(bs_b + off1, b1 + k0);
    };

    issue(0, 0);
    int buf = 0;
    for (int k0 = 0; k0 < K; k0 += 32) {
        wait_async();          // our DMA into buf is complete
        __syncthreads();       // everyone's DMA done; prior reads of buf^1 done
        if (k0 + 32 < K) issue(k0 + 32, buf ^ 1);

        const unsigned short* As = smem + buf * 8192;
        const unsigned short* Bs = As + 4096;
        v16bf af0 = load_frag(As + (wm + lr)      * 32 + kb);
        v16bf af1 = load_frag(As + (wm + 16 + lr) * 32 + kb);
        v16bf bf0 = load_frag(Bs + (wn + lr)      * 32 + kb);
        v16bf bf1 = load_frag(Bs + (wn + 16 + lr) * 32 + kb);
        v16bf bf2 = load_frag(Bs + (wn + 32 + lr) * 32 + kb);
        v16bf bf3 = load_frag(Bs + (wn + 48 + lr) * 32 + kb);
        acc[0] = __builtin_amdgcn_wmma_f32_16x16x32_bf16(false, af0, false, bf0, (short)0, acc[0], false, false);
        acc[4] = __builtin_amdgcn_wmma_f32_16x16x32_bf16(false, af1, false, bf0, (short)0, acc[4], false, false);
        acc[1] = __builtin_amdgcn_wmma_f32_16x16x32_bf16(false, af0, false, bf1, (short)0, acc[1], false, false);
        acc[5] = __builtin_amdgcn_wmma_f32_16x16x32_bf16(false, af1, false, bf1, (short)0, acc[5], false, false);
        acc[2] = __builtin_amdgcn_wmma_f32_16x16x32_bf16(false, af0, false, bf2, (short)0, acc[2], false, false);
        acc[6] = __builtin_amdgcn_wmma_f32_16x16x32_bf16(false, af1, false, bf2, (short)0, acc[6], false, false);
        acc[3] = __builtin_amdgcn_wmma_f32_16x16x32_bf16(false, af0, false, bf3, (short)0, acc[3], false, false);
        acc[7] = __builtin_amdgcn_wmma_f32_16x16x32_bf16(false, af1, false, bf3, (short)0, acc[7], false, false);
        buf ^= 1;
    }
}

// ---------------------------------------------------------------------------
// Unified WMMA GEMM:  C = A[M,K] * Bt[N,K]^T  (+bias) (+relu) (+residual)
// Outputs: fp32 Cf, bf16 Cb, bf16-transposed CbT.  blockIdx.z batches.
// ---------------------------------------------------------------------------
__global__ __launch_bounds__(256)
void gemm_bf16(const unsigned short* __restrict__ A,  int lda, long sA,
               const unsigned short* __restrict__ Bt, int ldb, long sB,
               int M, int N, int K,
               const float* __restrict__ bias,
               const float* __restrict__ residual,
               float*          __restrict__ Cf,  long sCf,
               unsigned short* __restrict__ Cb,  long sCb,
               unsigned short* __restrict__ CbT, int ldct, long sCt,
               int relu)
{
    __shared__ __align__(16) unsigned short smem[2 * 8192];

    const int m0 = blockIdx.x * 128;
    const int n0 = blockIdx.y * 128;
    const int z  = blockIdx.z;
    A  += (size_t)z * sA;
    Bt += (size_t)z * sB;

    const int lane = threadIdx.x & 31;
    const int wave = threadIdx.x >> 5;
    const int lr   = lane & 15;
    const int wm   = (wave & 3) * 32;
    const int wn   = (wave >> 2) * 64;

    v8f acc[8];
    #pragma unroll
    for (int i = 0; i < 8; ++i)
        #pragma unroll
        for (int e = 0; e < 8; ++e) acc[i][e] = 0.0f;

    wmma_tile_128x128(A, lda, M, m0, Bt, ldb, N, n0, K, smem, acc);

    // Epilogue. C layout: lane l, vgpr r -> (m = r + 8*(l>=16), n = l&15)
    #pragma unroll
    for (int mi = 0; mi < 2; ++mi) {
        #pragma unroll
        for (int nt = 0; nt < 4; ++nt) {
            int gn = n0 + wn + nt * 16 + lr;
            if (gn >= N) continue;
            float bv = bias ? bias[gn] : 0.0f;
            #pragma unroll
            for (int r = 0; r < 8; ++r) {
                int gm = m0 + wm + mi * 16 + r + ((lane >> 4) << 3);
                if (gm >= M) continue;
                float v = acc[mi * 4 + nt][r] + bv;
                if (relu && v < 0.0f) v = 0.0f;
                if (residual) v += residual[(size_t)gm * N + gn];
                if (Cf)  Cf [(size_t)z * sCf + (size_t)gm * N + gn] = v;
                if (Cb)  Cb [(size_t)z * sCb + (size_t)gm * N + gn] = f2bf(v);
                if (CbT) CbT[(size_t)z * sCt + (size_t)gn * ldct + gm] = f2bf(v);
            }
        }
    }
}

// ---------------------------------------------------------------------------
// Attention scores: scores[b,i,j] = (q_i.k_j)*scale^2 + qrel[b,i,clip(i-j,10)]*scale
// masked (j>i) -> -1e9.
// ---------------------------------------------------------------------------
__global__ __launch_bounds__(256)
void attn_scores(const unsigned short* __restrict__ qb,
                 const unsigned short* __restrict__ kbuf,
                 const float* __restrict__ qrel,
                 float* __restrict__ scores,
                 float scale)
{
    const int m0 = blockIdx.x * 128;
    const int n0 = blockIdx.y * 128;
    const int b  = blockIdx.z;
    float* sc = scores + (size_t)b * S_ * S_;

    if (n0 >= m0 + 128) {           // tile entirely above the diagonal
        for (int idx = threadIdx.x; idx < 128 * 128; idx += 256) {
            int mm = idx >> 7, nn = idx & 127;
            sc[(size_t)(m0 + mm) * S_ + n0 + nn] = -1e9f;
        }
        return;
    }

    __shared__ __align__(16) unsigned short smem[2 * 8192];

    const unsigned short* A  = qb   + (size_t)b * S_ * D_;
    const unsigned short* Bt = kbuf + (size_t)b * S_ * D_;

    const int lane = threadIdx.x & 31;
    const int wave = threadIdx.x >> 5;
    const int lr   = lane & 15;
    const int wm   = (wave & 3) * 32;
    const int wn   = (wave >> 2) * 64;

    v8f acc[8];
    #pragma unroll
    for (int i = 0; i < 8; ++i)
        #pragma unroll
        for (int e = 0; e < 8; ++e) acc[i][e] = 0.0f;

    wmma_tile_128x128(A, D_, S_, m0, Bt, D_, S_, n0, D_, smem, acc);

    const float s2 = scale * scale;
    #pragma unroll
    for (int mi = 0; mi < 2; ++mi) {
        #pragma unroll
        for (int nt = 0; nt < 4; ++nt) {
            int j = n0 + wn + nt * 16 + lr;
            #pragma unroll
            for (int r = 0; r < 8; ++r) {
                int i = m0 + wm + mi * 16 + r + ((lane >> 4) << 3);
                float v;
                if (j > i) {
                    v = -1e9f;
                } else {
                    int d = i - j; if (d > SPAN_) d = SPAN_;
                    v = acc[mi * 4 + nt][r] * s2 +
                        qrel[((size_t)b * S_ + i) * 16 + d] * scale;
                }
                sc[(size_t)i * S_ + j] = v;
            }
        }
    }
}

// ---------------------------------------------------------------------------
// Row softmax over S=512; writes bf16 probabilities.
// ---------------------------------------------------------------------------
__global__ __launch_bounds__(256)
void softmax_rows(const float* __restrict__ scores, unsigned short* __restrict__ P)
{
    const size_t row = blockIdx.x;
    const float* s = scores + row * S_;
    unsigned short* p = P + row * S_;
    __shared__ float red[256];
    int t = threadIdx.x;
    float a = s[t], b = s[t + 256];

    red[t] = fmaxf(a, b); __syncthreads();
    for (int off = 128; off > 0; off >>= 1) {
        if (t < off) red[t] = fmaxf(red[t], red[t + off]);
        __syncthreads();
    }
    float mx = red[0]; __syncthreads();

    float ea = __expf(a - mx), eb = __expf(b - mx);
    red[t] = ea + eb; __syncthreads();
    for (int off = 128; off > 0; off >>= 1) {
        if (t < off) red[t] += red[t + off];
        __syncthreads();
    }
    float inv = 1.0f / red[0];
    p[t]       = f2bf(ea * inv);
    p[t + 256] = f2bf(eb * inv);
}

// ---------------------------------------------------------------------------
// LayerNorm over D=512; writes fp32 h + bf16 copy.
// ---------------------------------------------------------------------------
__global__ __launch_bounds__(256)
void layernorm_rows(const float* __restrict__ X,
                    const float* __restrict__ g, const float* __restrict__ bta,
                    float* __restrict__ Yf, unsigned short* __restrict__ Yb)
{
    const size_t row = blockIdx.x;
    const float* x = X + row * D_;
    __shared__ float red[256];
    int t = threadIdx.x;
    float a = x[t], b = x[t + 256];

    red[t] = a + b; __syncthreads();
    for (int off = 128; off > 0; off >>= 1) {
        if (t < off) red[t] += red[t + off];
        __syncthreads();
    }
    float mu = red[0] * (1.0f / D_); __syncthreads();

    float da = a - mu, db = b - mu;
    red[t] = da * da + db * db; __syncthreads();
    for (int off = 128; off > 0; off >>= 1) {
        if (t < off) red[t] += red[t + off];
        __syncthreads();
    }
    float rs = rsqrtf(red[0] * (1.0f / D_) + 1e-5f);

    float ya = da * rs * g[t]       + bta[t];
    float yb = db * rs * g[t + 256] + bta[t + 256];
    Yf[row * D_ + t]       = ya;
    Yf[row * D_ + t + 256] = yb;
    Yb[row * D_ + t]       = f2bf(ya);
    Yb[row * D_ + t + 256] = f2bf(yb);
}

// ---------------------------------------------------------------------------
// Token embedding gather -> fp32 h (d_out) + bf16 copy.
// ---------------------------------------------------------------------------
__global__ __launch_bounds__(256)
void embed_rows(const int* __restrict__ x, const float* __restrict__ tok,
                float* __restrict__ hf, unsigned short* __restrict__ hb)
{
    const size_t row = blockIdx.x;
    const float* e = tok + (size_t)x[row] * D_;
    int t = threadIdx.x;
    float a = e[t], b = e[t + 256];
    hf[row * D_ + t]       = a;
    hf[row * D_ + t + 256] = b;
    hb[row * D_ + t]       = f2bf(a);
    hb[row * D_ + t + 256] = f2bf(b);
}

// ---------------------------------------------------------------------------
// Weight convert + transpose: W[K,N] fp32 -> Wt[N,K] bf16 bits.
// ---------------------------------------------------------------------------
__global__ __launch_bounds__(256)
void wtrans(const float* __restrict__ W, unsigned short* __restrict__ Wt,
            int K, int N)
{
    int idx = blockIdx.x * 256 + threadIdx.x;
    if (idx >= K * N) return;
    int k = idx / N, n = idx - k * N;
    Wt[(size_t)n * K + k] = f2bf(W[idx]);
}

// rel_emb rows 0..10 -> relb[16][512] bf16 (rows 11..15 clamp-padded, unused)
__global__ __launch_bounds__(256)
void relconv(const float* __restrict__ rel, unsigned short* __restrict__ relb)
{
    int idx = blockIdx.x * 256 + threadIdx.x;   // 16*512
    int n = idx >> 9, k = idx & 511;
    int rn = n > SPAN_ ? SPAN_ : n;
    relb[idx] = f2bf(rel[(size_t)rn * D_ + k]);
}

// ---------------------------------------------------------------------------
// Host driver
// ---------------------------------------------------------------------------
extern "C" void kernel_launch(void* const* d_in, const int* in_sizes, int n_in,
                              void* d_out, int out_size, void* d_ws, size_t ws_size,
                              hipStream_t stream)
{
    (void)in_sizes; (void)n_in; (void)out_size; (void)ws_size;

    const int*   x    = (const int*)  d_in[0];
    const float* tok  = (const float*)d_in[1];
    const float* rel  = (const float*)d_in[2];
    const float* Wq   = (const float*)d_in[3];
    const float* bq   = (const float*)d_in[4];
    const float* Wk   = (const float*)d_in[5];
    const float* bk   = (const float*)d_in[6];
    const float* Wv   = (const float*)d_in[7];
    const float* bv   = (const float*)d_in[8];
    const float* Wo   = (const float*)d_in[9];
    const float* bo   = (const float*)d_in[10];
    const float* W1   = (const float*)d_in[11];
    const float* b1   = (const float*)d_in[12];
    const float* W2   = (const float*)d_in[13];
    const float* b2   = (const float*)d_in[14];
    const float* ln1g = (const float*)d_in[15];
    const float* ln1b = (const float*)d_in[16];
    const float* ln2g = (const float*)d_in[17];
    const float* ln2b = (const float*)d_in[18];

    float* h = (float*)d_out;                 // fp32 residual stream == output
    const size_t R = (size_t)B_ * S_;         // 8192 rows

    char* wp = (char*)d_ws;
    auto alloc = [&](size_t bytes) {
        char* p = wp; wp += (bytes + 255) & ~(size_t)255; return p;
    };
    unsigned short* hb    = (unsigned short*)alloc(R * D_ * 2);
    unsigned short* qb    = (unsigned short*)alloc(R * D_ * 2);
    unsigned short* kbuf  = (unsigned short*)alloc(R * D_ * 2);
    unsigned short* vT    = (unsigned short*)alloc((size_t)B_ * D_ * S_ * 2);
    unsigned short* ctx   = (unsigned short*)alloc(R * D_ * 2);
    unsigned short* P     = (unsigned short*)alloc((size_t)B_ * S_ * S_ * 2);
    unsigned short* fbuf  = (unsigned short*)alloc(R * FF_ * 2);
    float*          qrel  = (float*)         alloc(R * 16 * 4);
    float*          sc    = (float*)         alloc((size_t)B_ * S_ * S_ * 4);
    float*          rbuf  = (float*)         alloc(R * D_ * 4);
    unsigned short* wq_t  = (unsigned short*)alloc((size_t)D_ * D_ * 2);
    unsigned short* wk_t  = (unsigned short*)alloc((size_t)D_ * D_ * 2);
    unsigned short* wv_t  = (unsigned short*)alloc((size_t)D_ * D_ * 2);
    unsigned short* wo_t  = (unsigned short*)alloc((size_t)D_ * D_ * 2);
    unsigned short* w1_t  = (unsigned short*)alloc((size_t)D_ * FF_ * 2);
    unsigned short* w2_t  = (unsigned short*)alloc((size_t)FF_ * D_ * 2);
    unsigned short* relb  = (unsigned short*)alloc(16 * D_ * 2);

    const float scale = 1.0f / sqrtf((float)D_);

    auto gemm = [&](const unsigned short* A, int lda, long sA,
                    const unsigned short* Bt, int ldb, long sB,
                    int M, int N, int K, int Z,
                    const float* bias, const float* resid,
                    float* Cf, long sCf,
                    unsigned short* Cb, long sCb,
                    unsigned short* CbT, int ldct, long sCt, int relu) {
        dim3 g((M + 127) / 128, (N + 127) / 128, Z);
        gemm_bf16<<<g, dim3(256), 0, stream>>>(A, lda, sA, Bt, ldb, sB,
                                               M, N, K, bias, resid,
                                               Cf, sCf, Cb, sCb, CbT, ldct, sCt,
                                               relu);
    };

    relconv<<<(16 * D_) / 256, 256, 0, stream>>>(rel, relb);
    embed_rows<<<(int)R, 256, 0, stream>>>(x, tok, h, hb);

    for (int l = 0; l < L_; ++l) {
        const int nDD = (D_ * D_ + 255) / 256;
        wtrans<<<nDD, 256, 0, stream>>>(Wq + (size_t)l * D_ * D_, wq_t, D_, D_);
        wtrans<<<nDD, 256, 0, stream>>>(Wk + (size_t)l * D_ * D_, wk_t, D_, D_);
        wtrans<<<nDD, 256, 0, stream>>>(Wv + (size_t)l * D_ * D_, wv_t, D_, D_);
        wtrans<<<nDD, 256, 0, stream>>>(Wo + (size_t)l * D_ * D_, wo_t, D_, D_);
        wtrans<<<(D_ * FF_ + 255) / 256, 256, 0, stream>>>(W1 + (size_t)l * D_ * FF_, w1_t, D_, FF_);
        wtrans<<<(FF_ * D_ + 255) / 256, 256, 0, stream>>>(W2 + (size_t)l * FF_ * D_, w2_t, FF_, D_);

        // q, k: bf16 only.  v: written transposed per batch -> vT[b][d][j].
        gemm(hb, D_, 0, wq_t, D_, 0, (int)R, D_, D_, 1,
             bq + (size_t)l * D_, nullptr, nullptr, 0, qb, 0, nullptr, 0, 0, 0);
        gemm(hb, D_, 0, wk_t, D_, 0, (int)R, D_, D_, 1,
             bk + (size_t)l * D_, nullptr, nullptr, 0, kbuf, 0, nullptr, 0, 0, 0);
        gemm(hb, D_, (long)S_ * D_, wv_t, D_, 0, S_, D_, D_, B_,
             bv + (size_t)l * D_, nullptr, nullptr, 0, nullptr, 0,
             vT, S_, (long)D_ * S_, 0);

        // qrel[b,i,n] = q . rel_emb[n]   (N=16 tile, only n<=10 consumed)
        gemm(qb, D_, 0, relb, D_, 0, (int)R, 16, D_, 1,
             nullptr, nullptr, qrel, 0, nullptr, 0, nullptr, 0, 0, 0);

        attn_scores<<<dim3(S_ / 128, S_ / 128, B_), dim3(256), 0, stream>>>(
            qb, kbuf, qrel, sc, scale);

        softmax_rows<<<(int)R, 256, 0, stream>>>(sc, P);

        // ctx[b] = P[b] @ V[b]   (B operand = vT, same loader as weights)
        gemm(P, S_, (long)S_ * S_, vT, S_, (long)D_ * S_, S_, D_, S_, B_,
             nullptr, nullptr, nullptr, 0, ctx, (long)S_ * D_, nullptr, 0, 0, 0);

        // attn out + residual -> rbuf; LN1 -> h (d_out) + hb
        gemm(ctx, D_, 0, wo_t, D_, 0, (int)R, D_, D_, 1,
             bo + (size_t)l * D_, h, rbuf, 0, nullptr, 0, nullptr, 0, 0, 0);
        layernorm_rows<<<(int)R, 256, 0, stream>>>(
            rbuf, ln1g + (size_t)l * D_, ln1b + (size_t)l * D_, h, hb);

        // FF
        gemm(hb, D_, 0, w1_t, D_, 0, (int)R, FF_, D_, 1,
             b1 + (size_t)l * FF_, nullptr, nullptr, 0, fbuf, 0, nullptr, 0, 0, 1);
        gemm(fbuf, FF_, 0, w2_t, FF_, 0, (int)R, D_, FF_, 1,
             b2 + (size_t)l * D_, h, rbuf, 0, nullptr, 0, nullptr, 0, 0, 0);
        layernorm_rows<<<(int)R, 256, 0, stream>>>(
            rbuf, ln2g + (size_t)l * D_, ln2b + (size_t)l * D_, h, hb);
    }
}
